// FastCapsNetMulti_3238405341384
// MI455X (gfx1250) — compile-verified, matrix operand order checked by reading, not run on hardware
//
#include <hip/hip_runtime.h>
#include <hip/hip_bf16.h>

// MI455X (gfx1250) CapsNet pipeline:
//  conv (implicit GEMM, bf16 WMMA, f32 accum) -> in-register squash -> bf16 caps
//  routing einsum as 64x160x65536 GEMM (bf16 WMMA, async LDS copy for caps)
//  finalize: reduce 256 partials, squash, logits.

typedef __attribute__((ext_vector_type(16))) __bf16 v16bf;
typedef __attribute__((ext_vector_type(8)))  __bf16 v8bf;
typedef __attribute__((ext_vector_type(8)))  float  v8f;

#define BATCH   64
#define CAPS_K  65536           // 8192 capsules * 8 dims
#define NCOLS   160             // 10 classes * 16 dims
#define NPART   256             // routing K-split blocks (256 K each)
#define KTAB_MAX 1312           // 41 chunks * 32 (Cin=144)

// ---------------------------------------------------------------------------
// Conv as implicit GEMM: M=32 out-ch, N=128 positions/block, K=Cin*9.
// One block = (batch b, 128-position tile). 256 threads = 8 waves.
// Per-k (c,dy,dx) decomposition precomputed once per block -> no idiv in the
// hot staging loop. Epilogue: each lane's 8 accumulators form one capsule
// vector (d = vgpr index); bias+squash in-register, 16B bf16 store.
// ---------------------------------------------------------------------------
__global__ __launch_bounds__(256) void caps_conv_wmma(
    const float* __restrict__ x,     // [B][Cin][32][32]
    const float* __restrict__ wgt,   // [32][Cin*9]
    const float* __restrict__ bias,  // [32]
    int Cin, int n_offset,
    __bf16* __restrict__ caps)       // [B][65536] bf16
{
    const int b    = blockIdx.y;
    const int p0   = blockIdx.x * 128;
    const int tid  = threadIdx.x;
    const int lane = tid & 31;
    const int wave = tid >> 5;
    const int Ktot = Cin * 9;
    const int nChunks = (Ktot + 31) >> 5;
    const float* __restrict__ xb = x + (size_t)b * Cin * 1024;

    __shared__ __bf16 Asl[32][32];        // [out_ch][k]
    __shared__ __bf16 Bsl[32][128 + 8];   // [k][position], padded
    __shared__ int         koff[KTAB_MAX];     // c*1024 + dy*32 + dx
    __shared__ signed char kdy[KTAB_MAX];      // 100 => k out of range
    __shared__ signed char kdx[KTAB_MAX];

    // one-time k -> (c, dy, dx) tables (only place with integer division)
    for (int k = tid; k < nChunks * 32; k += 256) {
        if (k < Ktot) {
            int c = k / 9, rs = k - 9 * c;
            int dy = rs / 3 - 1, dx = rs - 3 * (rs / 3) - 1;
            koff[k] = c * 1024 + dy * 32 + dx;
            kdy[k] = (signed char)dy;
            kdx[k] = (signed char)dx;
        } else {
            koff[k] = 0; kdy[k] = 100; kdx[k] = 100;  // forces OOB -> 0
        }
    }
    __syncthreads();

    v8f acc0 = {};   // out_ch 0..15
    v8f acc1 = {};   // out_ch 16..31

    for (int ck = 0; ck < nChunks; ++ck) {
        const int k0 = ck << 5;
        // stage weight slab (32 ch x 32 k), zero-pad K tail
        for (int i = tid; i < 32 * 32; i += 256) {
            int m = i >> 5, k = i & 31, kg = k0 + k;
            float v = (kg < Ktot) ? wgt[m * Ktot + kg] : 0.0f;
            Asl[m][k] = (__bf16)v;
        }
        // stage im2col slab (32 k x 128 positions) via tables
        for (int i = tid; i < 32 * 128; i += 256) {
            int k = i >> 7, n = i & 127, kg = k0 + k;
            int p = p0 + n;
            int y  = (p >> 5) + kdy[kg];
            int xx = (p & 31) + kdx[kg];
            float v = 0.0f;
            if ((unsigned)y < 32u && (unsigned)xx < 32u)
                v = xb[koff[kg] + p];
            Bsl[k][n] = (__bf16)v;
        }
        __syncthreads();

        // fragments per ISA 16-bit A/B lane layout
        const int row = lane & 15;
        const int kb  = (lane < 16) ? 0 : 8;
        const int col = wave * 16 + (lane & 15);
        v16bf a0, a1, bb;
        #pragma unroll
        for (int e = 0; e < 8; ++e) {
            a0[e]     = Asl[row][kb + e];
            a0[e + 8] = Asl[row][16 + kb + e];
            a1[e]     = Asl[16 + row][kb + e];
            a1[e + 8] = Asl[16 + row][16 + kb + e];
            bb[e]     = Bsl[kb + e][col];
            bb[e + 8] = Bsl[16 + kb + e][col];
        }
        acc0 = __builtin_amdgcn_wmma_f32_16x16x32_bf16(false, a0, false, bb,
                                                       (short)0, acc0, false, false);
        acc1 = __builtin_amdgcn_wmma_f32_16x16x32_bf16(false, a1, false, bb,
                                                       (short)0, acc1, false, false);
        __syncthreads();
    }

    // epilogue: lane holds capsule (cap, position), vgpr r = dim d
    const int p  = p0 + wave * 16 + (lane & 15);
    const int hi = (lane >= 16) ? 1 : 0;
    #pragma unroll
    for (int mt = 0; mt < 2; ++mt) {
        v8f acc = mt ? acc1 : acc0;
        int cap = mt * 2 + hi;                 // out_ch = cap*8 + d
        float v[8], sq = 0.0f;
        #pragma unroll
        for (int r = 0; r < 8; ++r) {
            v[r] = acc[r] + bias[cap * 8 + r];
            sq += v[r] * v[r];
        }
        float scale = sq / (1.0f + sq) / sqrtf(sq + 1e-8f);
        v8bf outv;
        #pragma unroll
        for (int r = 0; r < 8; ++r) outv[r] = (__bf16)(v[r] * scale);
        int n = n_offset + p * 4 + cap;
        *(v8bf*)(caps + (size_t)b * CAPS_K + (size_t)n * 8) = outv;
    }
}

// ---------------------------------------------------------------------------
// Routing einsum as GEMM: [64 x 65536] (caps, bf16) x [65536 x 160] (W).
// 256 blocks split K (256 each = 8 WMMA steps). W slab per step is 5120
// CONTIGUOUS floats (k = n*8+d matches W's [n][c][j][d] layout). caps slab is
// a raw bf16 copy -> moved with GLOBAL_LOAD_ASYNC_TO_LDS_B128 (ASYNCcnt).
// Output tile grid 4(M) x 10(N); 8 waves x 5 tiles. Partials -> ws.
// ---------------------------------------------------------------------------
__global__ __launch_bounds__(256) void caps_route_wmma(
    const __bf16* __restrict__ caps,   // [64][65536] bf16
    const float*  __restrict__ Wg,     // [8192][10][16][8] f32
    float* __restrict__ partial)       // [NPART][64][160] f32
{
    const int blk  = blockIdx.x;
    const int tid  = threadIdx.x;
    const int lane = tid & 31;
    const int wave = tid >> 5;
    const int kBase = blk * 256;

    __shared__ __bf16 Wsl[32][NCOLS + 8];  // [k_local][col]
    __shared__ __bf16 Csl[64][32];         // [batch][k_local]

    const int m0    = (wave & 3) * 16;     // batch tile
    const int nBase = (wave >> 2) * 5;     // first of 5 N-tiles

    // per-thread async-copy coordinates: 64 rows x 64 B, one B128 per thread
    const int crow = tid >> 2;
    const int cseg = (tid & 3) * 8;        // bf16 elements (16 B)
    unsigned ldsCdst = (unsigned)(uintptr_t)&Csl[crow][cseg];

    v8f acc[5] = {{}, {}, {}, {}, {}};

    for (int step = 0; step < 8; ++step) {
        const int k0 = kBase + step * 32;
        const float* wsrc = Wg + (size_t)(k0 >> 3) * 1280;  // 4 capsules' W rows

        // kick off caps slab: 256 lanes x 16 B async global->LDS
        {
            const __bf16* gsrc = caps + (size_t)crow * CAPS_K + k0 + cseg;
            asm volatile("global_load_async_to_lds_b128 %0, %1, off"
                         :: "v"(ldsCdst), "v"(gsrc) : "memory");
        }

        // prefetch next step's contiguous 20KB W slab into GL2
        if (step < 7 && tid < 160)
            __builtin_prefetch((const char*)(wsrc + 5120) + tid * 128, 0, 1);

        // stage W slab: permute [n_local][c][j][d] -> [k_local][c*16+j] bf16
        #pragma unroll
        for (int nl = 0; nl < 4; ++nl) {
            const float* src = wsrc + nl * 1280;
            for (int i = tid; i < 1280; i += 256) {
                int c = i >> 7, j = (i >> 3) & 15, d = i & 7;
                Wsl[nl * 8 + d][c * 16 + j] = (__bf16)src[i];
            }
        }
        asm volatile("s_wait_asynccnt 0x0" ::: "memory");
        __syncthreads();

        const int arow = m0 + (lane & 15);
        const int kb   = (lane < 16) ? 0 : 8;
        v16bf a;
        #pragma unroll
        for (int e = 0; e < 8; ++e) {
            a[e]     = Csl[arow][kb + e];
            a[e + 8] = Csl[arow][16 + kb + e];
        }
        #pragma unroll
        for (int t = 0; t < 5; ++t) {
            const int col = (nBase + t) * 16 + (lane & 15);
            v16bf bb;
            #pragma unroll
            for (int e = 0; e < 8; ++e) {
                bb[e]     = Wsl[kb + e][col];
                bb[e + 8] = Wsl[16 + kb + e][col];
            }
            acc[t] = __builtin_amdgcn_wmma_f32_16x16x32_bf16(false, a, false, bb,
                                                             (short)0, acc[t], false, false);
        }
        __syncthreads();
    }

    // write per-block partial (no atomics; deterministic)
    float* pout = partial + (size_t)blk * (BATCH * NCOLS);
    const int hi = (lane >= 16) ? 8 : 0;
    #pragma unroll
    for (int t = 0; t < 5; ++t) {
        const int col = (nBase + t) * 16 + (lane & 15);
        #pragma unroll
        for (int r = 0; r < 8; ++r)
            pout[(size_t)(m0 + r + hi) * NCOLS + col] = acc[t][r];
    }
}

// ---------------------------------------------------------------------------
// Finalize: sum NPART partials, /N, squash, logits = |v|.
// ---------------------------------------------------------------------------
__global__ __launch_bounds__(128) void caps_logits(
    const float* __restrict__ partial, float* __restrict__ out)
{
    int t = blockIdx.x * 128 + threadIdx.x;
    if (t >= BATCH * 10) return;
    int b = t / 10, c = t % 10;
    float sq = 0.0f;
    for (int j = 0; j < 16; ++j) {
        float s = 0.0f;
        for (int p = 0; p < NPART; ++p)
            s += partial[(size_t)p * (BATCH * NCOLS) + (size_t)b * NCOLS + c * 16 + j];
        s *= (1.0f / 8192.0f);
        sq += s * s;
    }
    // |v| = sqrt(sq) * sq / ((1+sq) * sqrt(sq + 1e-8))
    out[t] = sqrtf(sq) * sq / ((1.0f + sq) * sqrtf(sq + 1e-8f));
}

extern "C" void kernel_launch(void* const* d_in, const int* in_sizes, int n_in,
                              void* d_out, int out_size, void* d_ws, size_t ws_size,
                              hipStream_t stream) {
    (void)in_sizes; (void)n_in; (void)out_size; (void)ws_size;
    const float* x_hsi = (const float*)d_in[0];   // [64,144,32,32]
    const float* x_sar = (const float*)d_in[1];   // [64,4,32,32]
    const float* wh    = (const float*)d_in[2];   // [32,144,3,3]
    const float* bh    = (const float*)d_in[3];   // [32]
    const float* ws    = (const float*)d_in[4];   // [32,4,3,3]
    const float* bs    = (const float*)d_in[5];   // [32]
    const float* Wbig  = (const float*)d_in[6];   // [8192,10,16,8]
    float* out = (float*)d_out;                   // [64,10]

    __bf16* caps   = (__bf16*)d_ws;                                     // 8 MB
    float* partial = (float*)((char*)d_ws +
                              (size_t)BATCH * CAPS_K * sizeof(__bf16)); // 10.5 MB

    caps_conv_wmma<<<dim3(8, BATCH), 256, 0, stream>>>(x_hsi, wh, bh, 144, 0,    caps);
    caps_conv_wmma<<<dim3(8, BATCH), 256, 0, stream>>>(x_sar, ws, bs, 4,   4096, caps);
    caps_route_wmma<<<NPART, 256, 0, stream>>>(caps, Wbig, partial);
    caps_logits<<<5, 128, 0, stream>>>(partial, out);
}
